// FunnelGNN_old_63574105915954
// MI455X (gfx1250) — compile-verified, matrix operand order checked by reading, not run on hardware
//
#include <hip/hip_runtime.h>
#include <hip/hip_bf16.h>

// ---------------------------------------------------------------------------
// FunnelGNN forward for MI455X (gfx1250, wave32).
// GEMMs run through v_wmma_f32_16x16x32_f16; edge scatter vectorized 4-wide
// with shift/mask indexing (all channel counts are powers of two).
// ---------------------------------------------------------------------------

typedef __attribute__((ext_vector_type(16))) _Float16 v16h;
typedef __attribute__((ext_vector_type(8)))  _Float16 v8h;
typedef __attribute__((ext_vector_type(8)))  float    v8f;

#define EPS 1e-5f
#define NEG_SLOPE 0.01f

// ---------------- utility kernels ----------------

__global__ void k_zero_f32(float* p, size_t n) {
    size_t i = (size_t)blockIdx.x * blockDim.x + threadIdx.x;
    size_t stride = (size_t)gridDim.x * blockDim.x;
    for (; i < n; i += stride) p[i] = 0.0f;
}

// scatter-add, 4 channels per thread: agg[dst[e]*C + c..c+3] += h[src[e]*C + c..c+3]
// logCg = log2(C/4); total = E * (C/4)
__global__ void k_scatter_add4(const float* __restrict__ h,
                               const int* __restrict__ src,
                               const int* __restrict__ dst,
                               float* __restrict__ agg,
                               int logCg, int C, long total) {
    long i = (long)blockIdx.x * blockDim.x + threadIdx.x;
    long stride = (long)gridDim.x * blockDim.x;
    int cgMask = (1 << logCg) - 1;
    for (; i < total; i += stride) {
        long e = i >> logCg;
        int  c = (int)(i & cgMask) << 2;
        int  s = src[e];
        int  d = dst[e];
        const float4 v = *(const float4*)(h + (size_t)s * C + c);
        float* a = agg + (size_t)d * C + c;
        atomicAdd(a + 0, v.x);
        atomicAdd(a + 1, v.y);
        atomicAdd(a + 2, v.z);
        atomicAdd(a + 3, v.w);
    }
}

// convert f32 weight (dout x din, row major) -> f16 padded to Kp columns
__global__ void k_cvt_weight(const float* __restrict__ w, _Float16* __restrict__ w16,
                             int dout, int din, int logKp) {
    int Kp = 1 << logKp;
    int total = dout << logKp;
    int i = blockIdx.x * blockDim.x + threadIdx.x;
    int stride = gridDim.x * blockDim.x;
    for (; i < total; i += stride) {
        int r = i >> logKp, k = i & (Kp - 1);
        w16[i] = (_Float16)((k < din) ? w[r * din + k] : 0.0f);
    }
}

// convert agg + h activations (N x C) -> f16 padded to Kp (Kp = 1<<logKp)
__global__ void k_cvt_act(const float* __restrict__ a, const float* __restrict__ h,
                          _Float16* __restrict__ a16, _Float16* __restrict__ h16,
                          int N, int C, int logKp) {
    int Kp = 1 << logKp;
    size_t total = (size_t)N << logKp;
    size_t i = (size_t)blockIdx.x * blockDim.x + threadIdx.x;
    size_t stride = (size_t)gridDim.x * blockDim.x;
    for (; i < total; i += stride) {
        size_t n = i >> logKp;
        int k = (int)(i & (size_t)(Kp - 1));
        float av = 0.0f, hv = 0.0f;
        if (k < C) { av = a[n * C + k]; hv = h[n * C + k]; }
        a16[i] = (_Float16)av;
        h16[i] = (_Float16)hv;
    }
}

// ---------------- WMMA GEMM (fused GraphConv linear + bias + leaky) ----------------
// One wave computes one 16x16 output tile:
//   out[m,n] = leaky( sum_k A[m,k]*Wrel[n,k] + sum_k H[m,k]*Wroot[n,k] + brel[n] )
// 16-bit A/B tile layout per CDNA5 ISA: lanes 0-15 own K in [0..7]+[16..23],
// lanes 16-31 own K in [8..15]+[24..31]; rows/cols = lane%16.

__device__ __forceinline__ v16h load_tile16(const _Float16* __restrict__ base,
                                            int stride, int row, int kbase, int lane) {
    const _Float16* p = base + (size_t)row * stride + kbase + ((lane & 16) ? 8 : 0);
    const v8h* q = (const v8h*)p;      // 16B aligned: stride & kbase are 32-half multiples
    v8h lo = q[0];                     // K offsets +0..7
    v8h hi = q[2];                     // K offsets +16..23
    v16h r;
#pragma unroll
    for (int i = 0; i < 8; ++i) { r[i] = lo[i]; r[8 + i] = hi[i]; }
    return r;
}

__global__ void k_gemm_conv_wmma(const _Float16* __restrict__ A16,   // agg   [N x Kp]
                                 const _Float16* __restrict__ H16,   // h     [N x Kp]
                                 const _Float16* __restrict__ Wrel,  // [dout x Kp]
                                 const _Float16* __restrict__ Wroot, // [dout x Kp]
                                 const float* __restrict__ brel,
                                 float* __restrict__ out,            // [N x dout]
                                 int ntiles_m, int ntiles_n, int Kp, int dout) {
    int wave = (blockIdx.x * blockDim.x + threadIdx.x) >> 5;
    int lane = threadIdx.x & 31;
    int total = ntiles_m * ntiles_n;
    if (wave >= total) return;                 // wave-uniform: EXEC stays all-ones
    int mt = wave / ntiles_n;
    int nt = wave % ntiles_n;
    int mrow = mt * 16 + (lane & 15);
    int nrow = nt * 16 + (lane & 15);

    v8f c = {};
    for (int kb = 0; kb < Kp; kb += 32) {
        v16h a  = load_tile16(A16,  Kp, mrow, kb, lane);
        v16h br = load_tile16(Wrel, Kp, nrow, kb, lane);
        c = __builtin_amdgcn_wmma_f32_16x16x32_f16(false, a, false, br,
                                                   (short)0, c, false, false);
        v16h hm = load_tile16(H16,   Kp, mrow, kb, lane);
        v16h bo = load_tile16(Wroot, Kp, nrow, kb, lane);
        c = __builtin_amdgcn_wmma_f32_16x16x32_f16(false, hm, false, bo,
                                                   (short)0, c, false, false);
    }

    // C/D layout: lanes 0-15 -> N=lane, M=0..7 in v0..7; lanes 16-31 -> N=lane-16, M=8..15
    int n = nt * 16 + (lane & 15);
    int mbase = mt * 16 + ((lane & 16) ? 8 : 0);
    float bias = brel[n];
#pragma unroll
    for (int v = 0; v < 8; ++v) {
        float val = c[v] + bias;
        val = (val > 0.0f) ? val : NEG_SLOPE * val;   // leaky relu (pre-BN)
        out[(size_t)(mbase + v) * dout + n] = val;
    }
}

// ---------------- BatchNorm (training-mode stats over nodes) ----------------

__global__ void k_bn_stats(const float* __restrict__ h, float* __restrict__ sums,
                           int N, int C) {
    int c = blockIdx.y;
    __shared__ float ss[256];
    __shared__ float sq[256];
    float s = 0.0f, q = 0.0f;
    for (long i = (long)blockIdx.x * blockDim.x + threadIdx.x; i < N;
         i += (long)gridDim.x * blockDim.x) {
        float v = h[i * C + c];
        s += v; q += v * v;
    }
    ss[threadIdx.x] = s; sq[threadIdx.x] = q;
    __syncthreads();
    for (int o = 128; o > 0; o >>= 1) {
        if ((int)threadIdx.x < o) {
            ss[threadIdx.x] += ss[threadIdx.x + o];
            sq[threadIdx.x] += sq[threadIdx.x + o];
        }
        __syncthreads();
    }
    if (threadIdx.x == 0) {
        atomicAdd(&sums[c], ss[0]);
        atomicAdd(&sums[C + c], sq[0]);
    }
}

__global__ void k_bn_apply(float* __restrict__ h, const float* __restrict__ sums,
                           const float* __restrict__ gam, const float* __restrict__ bet,
                           int N, int C) {
    size_t total = (size_t)N * C;
    size_t i = (size_t)blockIdx.x * blockDim.x + threadIdx.x;
    size_t stride = (size_t)gridDim.x * blockDim.x;
    float invN = 1.0f / (float)N;
    for (; i < total; i += stride) {
        int c = (int)(i % C);
        float mu = sums[c] * invN;
        float var = sums[C + c] * invN - mu * mu;
        h[i] = (h[i] - mu) * rsqrtf(var + EPS) * gam[c] + bet[c];
    }
}

// ---------------- pool3: concat(max, mean, sum) per graph ----------------

__global__ void k_pool3(const float* __restrict__ h, float* __restrict__ z,
                        int C, int zoff, int zstride, int npg) {
    int g = blockIdx.x;
    int c = threadIdx.x;
    if (c >= C) return;
    const float* p = h + (size_t)g * npg * C + c;
    float s = 0.0f, m = -__builtin_inff();
    for (int i = 0; i < npg; ++i) {
        float v = p[(size_t)i * C];
        s += v;
        m = fmaxf(m, v);
    }
    float* zr = z + (size_t)g * zstride + zoff;
    zr[c] = m;
    zr[C + c] = s / (float)npg;
    zr[2 * C + c] = s;
}

// ---------------- classifier head + log_softmax ----------------

__global__ void k_head(const float* __restrict__ z,
                       const float* __restrict__ w1, const float* __restrict__ b1,
                       const float* __restrict__ w2, const float* __restrict__ b2,
                       float* __restrict__ out, int zdim, int fc) {
    __shared__ float zs[576];
    __shared__ float hid[128];
    int g = blockIdx.x;
    int t = threadIdx.x;
    for (int i = t; i < zdim; i += blockDim.x) zs[i] = z[(size_t)g * zdim + i];
    __syncthreads();
    if (t < fc) {
        float acc = b1[t];
        const float* wr = w1 + (size_t)t * zdim;
        for (int k = 0; k < zdim; ++k) acc += wr[k] * zs[k];
        hid[t] = acc;
    }
    __syncthreads();
    if (t == 0) {
        float o0 = b2[0], o1 = b2[1];
        for (int k = 0; k < fc; ++k) {
            o0 += w2[k] * hid[k];
            o1 += w2[fc + k] * hid[k];
        }
        float m = fmaxf(o0, o1);
        float lse = m + logf(expf(o0 - m) + expf(o1 - m));
        out[g * 2 + 0] = o0 - lse;
        out[g * 2 + 1] = o1 - lse;
    }
}

// ---------------- host launcher ----------------

static inline char* ws_take(char*& p, size_t bytes) {
    char* r = p;
    p += (bytes + 255) & ~(size_t)255;
    return r;
}

static inline int ilog2(int v) { int l = 0; while ((1 << l) < v) ++l; return l; }

extern "C" void kernel_launch(void* const* d_in, const int* in_sizes, int n_in,
                              void* d_out, int out_size, void* d_ws, size_t ws_size,
                              hipStream_t stream) {
    const int IN_DIM = 16, H = 32, FC = 128;
    const int Nn = in_sizes[0] / IN_DIM;      // 100000
    const int E  = in_sizes[1] / 2;           // 800000
    const int G  = out_size / 2;              // 1000
    const int npg = Nn / G;                   // 100
    const int ZDIM = 18 * H;                  // 576

    const float* x    = (const float*)d_in[0];
    const int*   src  = (const int*)d_in[1];          // edge_index[0]
    const int*   dst  = ((const int*)d_in[1]) + E;    // edge_index[1]
    const float* wlin1 = (const float*)d_in[19];
    const float* blin1 = (const float*)d_in[20];
    const float* wlin2 = (const float*)d_in[21];
    const float* blin2 = (const float*)d_in[22];
    float* out = (float*)d_out;

    // ---- workspace carve-up ----
    char* w = (char*)d_ws;
    float*     hA     = (float*)    ws_take(w, (size_t)Nn * 96 * sizeof(float));
    float*     hB     = (float*)    ws_take(w, (size_t)Nn * 96 * sizeof(float));
    float*     agg    = (float*)    ws_take(w, (size_t)Nn * 64 * sizeof(float));
    _Float16*  a16    = (_Float16*) ws_take(w, (size_t)Nn * 64 * sizeof(_Float16));
    _Float16*  h16    = (_Float16*) ws_take(w, (size_t)Nn * 64 * sizeof(_Float16));
    _Float16*  wrel16 = (_Float16*) ws_take(w, (size_t)96 * 64 * sizeof(_Float16));
    _Float16*  wroot16= (_Float16*) ws_take(w, (size_t)96 * 64 * sizeof(_Float16));
    float*     bns    = (float*)    ws_take(w, (size_t)2 * 96 * sizeof(float));
    float*     z      = (float*)    ws_take(w, (size_t)G * ZDIM * sizeof(float));
    (void)ws_size; (void)n_in;

    // per-layer config
    const int Cin_a[3]  = { IN_DIM, H,      2 * H };
    const int Kp_a[3]   = { 32,     32,     64    };
    const int Cout_a[3] = { H,      2 * H,  3 * H };
    const float* hin_a[3]  = { x,  hA, hB };
    float*       hout_a[3] = { hA, hB, hA };
    const int widx[3] = { 4, 9, 14 };   // w_rel, b_rel, w_root, gamma, beta blocks
    const int zoff_a[3] = { 0, 3 * H, 9 * H };

    for (int l = 0; l < 3; ++l) {
        const int Cin = Cin_a[l], Kp = Kp_a[l], Cout = Cout_a[l];
        const int logKp = ilog2(Kp);
        const float* wrel  = (const float*)d_in[widx[l] + 0];
        const float* brel  = (const float*)d_in[widx[l] + 1];
        const float* wroot = (const float*)d_in[widx[l] + 2];
        const float* gam   = (const float*)d_in[widx[l] + 3];
        const float* bet   = (const float*)d_in[widx[l] + 4];
        const float* hin = hin_a[l];
        float* hout = hout_a[l];

        // 1) zero aggregation buffer
        {
            size_t tot = (size_t)Nn * Cin;
            int blocks = (int)((tot + 255) / 256);
            k_zero_f32<<<blocks, 256, 0, stream>>>(agg, tot);
        }
        // 2) edge scatter-add (4 channels / thread, shift-mask indexing)
        {
            int logCg = ilog2(Cin / 4);
            long tot = (long)E << logCg;
            int blocks = (int)((tot + 255) / 256);
            k_scatter_add4<<<blocks, 256, 0, stream>>>(hin, src, dst, agg,
                                                       logCg, Cin, tot);
        }
        // 3) weight + activation conversion to padded f16
        {
            int tot = Cout * Kp;
            int blocks = (tot + 255) / 256;
            k_cvt_weight<<<blocks, 256, 0, stream>>>(wrel,  wrel16,  Cout, Cin, logKp);
            k_cvt_weight<<<blocks, 256, 0, stream>>>(wroot, wroot16, Cout, Cin, logKp);
            size_t atot = (size_t)Nn * Kp;
            int ablocks = (int)((atot + 255) / 256);
            k_cvt_act<<<ablocks, 256, 0, stream>>>(agg, hin, a16, h16, Nn, Cin, logKp);
        }
        // 4) fused WMMA GEMM: rel + root + bias + leaky
        {
            int tm = Nn / 16, tn = Cout / 16;
            long threads = (long)tm * tn * 32;
            int blocks = (int)((threads + 255) / 256);
            k_gemm_conv_wmma<<<blocks, 256, 0, stream>>>(a16, h16, wrel16, wroot16,
                                                         brel, hout, tm, tn, Kp, Cout);
        }
        // 5) batchnorm
        {
            k_zero_f32<<<1, 256, 0, stream>>>(bns, (size_t)2 * Cout);
            dim3 sgrid(64, Cout);
            k_bn_stats<<<sgrid, 256, 0, stream>>>(hout, bns, Nn, Cout);
            size_t tot = (size_t)Nn * Cout;
            int blocks = (int)((tot + 255) / 256);
            k_bn_apply<<<blocks, 256, 0, stream>>>(hout, bns, gam, bet, Nn, Cout);
        }
        // 6) pool3 into z
        k_pool3<<<G, 128, 0, stream>>>(hout, z, Cout, zoff_a[l], ZDIM, npg);
    }

    // classifier head + log_softmax
    k_head<<<G, 128, 0, stream>>>(z, wlin1, blin1, wlin2, blin2, out, ZDIM, FC);
}
